// AutoCorrelation_9655086481522
// MI455X (gfx1250) — compile-verified, matrix operand order checked by reading, not run on hardware
//
#include <hip/hip_runtime.h>
#include <hip/hip_bf16.h>

typedef __attribute__((ext_vector_type(16))) _Float16 v16h;
typedef __attribute__((ext_vector_type(8)))  float    v8f;
typedef __attribute__((ext_vector_type(4)))  float    f32x4;
typedef __attribute__((ext_vector_type(4)))  _Float16 h4;

#define LL   4096
#define BB   8
#define FEAT 512      // H*E
#define TOPK 8

// ---------------------------------------------------------------------------
// f32 -> f16 conversion (q,k packed as [B, L, 512] f16, row contiguous)
// ---------------------------------------------------------------------------
__global__ __launch_bounds__(256) void cvt_f16_kernel(const float* __restrict__ src,
                                                      _Float16* __restrict__ dst, int n4) {
    int i = blockIdx.x * 256 + threadIdx.x;
    if (i < n4) {
        f32x4 x = *(const f32x4*)(src + (size_t)i * 4);
        h4 y;
        y.x = (_Float16)x.x; y.y = (_Float16)x.y;
        y.z = (_Float16)x.z; y.w = (_Float16)x.w;
        *(h4*)(dst + (size_t)i * 4) = y;
    }
}

__global__ __launch_bounds__(256) void zero_kernel(float* __restrict__ p, int n) {
    int i = blockIdx.x * 256 + threadIdx.x;
    if (i < n) p[i] = 0.0f;
}

// ---------------------------------------------------------------------------
// Async staging: global f16 rows -> LDS tile via CDNA5 async-to-LDS path.
// 8 async b128 ops per thread-call (4 reps x {q,k}), tracked by ASYNCcnt.
// ---------------------------------------------------------------------------
__device__ __forceinline__ void stage_async(const _Float16* __restrict__ base_q,
                                            const _Float16* __restrict__ base_k,
                                            _Float16 (*qbuf)[72], _Float16 (*kbuf)[72],
                                            int rb, int cb, int kc, int t) {
    #pragma unroll
    for (int rep = 0; rep < 4; ++rep) {
        int cidx = rep * 512 + t;
        int row  = cidx >> 3;
        int c    = cidx & 7;
        unsigned qoff = (unsigned)(((rb + row) * FEAT + kc * 64 + c * 8) * 2);  // bytes
        unsigned koff = (unsigned)(((cb + row) * FEAT + kc * 64 + c * 8) * 2);
        unsigned ldq  = (unsigned)(uintptr_t)(&qbuf[row][c * 8]);               // LDS addr
        unsigned ldk  = (unsigned)(uintptr_t)(&kbuf[row][c * 8]);
        asm volatile("global_load_async_to_lds_b128 %0, %1, %2"
                     :: "v"(ldq), "v"(qoff), "s"(base_q) : "memory");
        asm volatile("global_load_async_to_lds_b128 %0, %1, %2"
                     :: "v"(ldk), "v"(koff), "s"(base_k) : "memory");
    }
}

// ---------------------------------------------------------------------------
// Gram super-tile kernel: per WG one 256x256 tile of G_b = Q_b K_b^T (K=512),
// computed with v_wmma_f32_16x16x32_f16, double-buffered async LDS staging,
// reduced along circular diagonals into mean_value[b, tau] (scaled by 1/512).
// Grid: B*16*16 blocks, 512 threads (16 waves).
// ---------------------------------------------------------------------------
__global__ __launch_bounds__(512) void gram_diag_kernel(const _Float16* __restrict__ qh,
                                                        const _Float16* __restrict__ kh,
                                                        float* __restrict__ meanv) {
    __shared__ __align__(16) _Float16 qs[2][256][72];   // 64 feats + 8 pad (stride 144B)
    __shared__ __align__(16) _Float16 ks[2][256][72];
    __shared__ float s_lag[512];                        // lags d = i-j+255 in [0,510]

    const int bx  = blockIdx.x;
    const int bat = bx >> 8;
    const int rb  = ((bx >> 4) & 15) << 8;              // q row base
    const int cb  = (bx & 15) << 8;                     // k row base

    const _Float16* qbase = qh + (size_t)bat * LL * FEAT;
    const _Float16* kbase = kh + (size_t)bat * LL * FEAT;

    const int t    = threadIdx.x;
    const int w    = t >> 5;        // wave id 0..15 -> i-block
    const int lane = t & 31;
    const int m    = lane & 15;
    const int g    = lane >> 4;

    s_lag[t & 511] = 0.0f;

    v8f acc[16] = {};               // 16 j-block accumulators (16x16 f32 tiles)

    // Prologue: stage chunk 0 into buffer 0.
    stage_async(qbase, kbase, qs[0], ks[0], rb, cb, 0, t);

    for (int kc = 0; kc < 8; ++kc) {                    // 8 K-chunks of 64 features
        const int cur = kc & 1;
        if (kc < 7) {
            // Stream next chunk into the other buffer while we compute this one.
            stage_async(qbase, kbase, qs[cur ^ 1], ks[cur ^ 1], rb, cb, kc + 1, t);
            asm volatile("s_wait_asynccnt 0x8" ::: "memory");   // stage(kc) complete
        } else {
            asm volatile("s_wait_asynccnt 0x0" ::: "memory");
        }
        __syncthreads();                                 // all waves' stage(kc) visible

        // A fragments for this wave's i-block (two K=32 slabs)
        union Frag { v16h h; f32x4 q[2]; };
        Frag a[2];
        const _Float16* ar = &qs[cur][w * 16 + m][0];
        a[0].q[0] = *(const f32x4*)(ar + 8 * g);
        a[0].q[1] = *(const f32x4*)(ar + 16 + 8 * g);
        a[1].q[0] = *(const f32x4*)(ar + 32 + 8 * g);
        a[1].q[1] = *(const f32x4*)(ar + 48 + 8 * g);

        #pragma unroll
        for (int s = 0; s < 2; ++s) {                    // K=32 slabs
            #pragma unroll
            for (int jb = 0; jb < 16; ++jb) {
                Frag bf;
                const _Float16* br = &ks[cur][jb * 16 + m][s * 32];  // lane m = B column
                bf.q[0] = *(const f32x4*)(br + 8 * g);
                bf.q[1] = *(const f32x4*)(br + 16 + 8 * g);
                acc[jb] = __builtin_amdgcn_wmma_f32_16x16x32_f16(
                    false, a[s].h, false, bf.h, (short)0, acc[jb], false, false);
            }
        }
        __syncthreads();                                 // compute(kc) done everywhere
    }

    // C layout: VGPR r, lanes 0-15 -> M=r,N=lane ; lanes 16-31 -> M=r+8,N=lane-16
    #pragma unroll
    for (int jb = 0; jb < 16; ++jb) {
        int jloc  = jb * 16 + m;
        int iloc0 = w * 16 + 8 * g;
        #pragma unroll
        for (int r = 0; r < 8; ++r) {
            int d = iloc0 + r - jloc + 255;              // 0..510
            atomicAdd(&s_lag[d], acc[jb][r]);
        }
    }
    __syncthreads();

    if (t < 511) {
        int tau = (rb - cb + t - 255) & (LL - 1);
        atomicAdd(meanv + bat * LL + tau, s_lag[t] * (1.0f / (float)FEAT));
    }
}

// ---------------------------------------------------------------------------
// Top-K over batch-mean of mean_value, then per-batch softmax of weights.
// Single workgroup.
// ---------------------------------------------------------------------------
__global__ __launch_bounds__(256) void topk_softmax_kernel(const float* __restrict__ meanv,
                                                           int* __restrict__ idx_out,
                                                           float* __restrict__ w_out) {
    __shared__ float s_avg[LL];
    __shared__ float s_val[256];
    __shared__ int   s_idx[256];
    __shared__ int   s_sel[TOPK];
    const int t = threadIdx.x;

    for (int i = t; i < LL; i += 256) {
        float s = 0.0f;
        #pragma unroll
        for (int bb = 0; bb < BB; ++bb) s += meanv[bb * LL + i];
        s_avg[i] = s;
    }
    __syncthreads();

    for (int k = 0; k < TOPK; ++k) {
        float best = -3.4e38f; int bi = 0;
        for (int i = t; i < LL; i += 256) {
            float v = s_avg[i];
            if (v > best) { best = v; bi = i; }
        }
        s_val[t] = best; s_idx[t] = bi;
        __syncthreads();
        for (int s = 128; s > 0; s >>= 1) {
            if (t < s && s_val[t + s] > s_val[t]) {
                s_val[t] = s_val[t + s]; s_idx[t] = s_idx[t + s];
            }
            __syncthreads();
        }
        if (t == 0) {
            s_sel[k] = s_idx[0];
            idx_out[k] = s_idx[0];
            s_avg[s_idx[0]] = -3.4e38f;
        }
        __syncthreads();
    }

    if (t < BB) {
        float wv[TOPK]; float mx = -3.4e38f;
        #pragma unroll
        for (int k = 0; k < TOPK; ++k) {
            wv[k] = meanv[t * LL + s_sel[k]];
            mx = fmaxf(mx, wv[k]);
        }
        float sum = 0.0f;
        #pragma unroll
        for (int k = 0; k < TOPK; ++k) { wv[k] = __expf(wv[k] - mx); sum += wv[k]; }
        float inv = 1.0f / sum;
        #pragma unroll
        for (int k = 0; k < TOPK; ++k) w_out[t * TOPK + k] = wv[k] * inv;
    }
}

// ---------------------------------------------------------------------------
// out[b,t,f] = sum_i w[b,i] * v[b,(t+idx_i)%L,f], vectorized float4 over f.
// ---------------------------------------------------------------------------
__global__ __launch_bounds__(256) void gather_agg_kernel(const float* __restrict__ v,
                                                         const int* __restrict__ idx,
                                                         const float* __restrict__ wsm,
                                                         float* __restrict__ out) {
    size_t gid = (size_t)blockIdx.x * 256 + threadIdx.x;   // over B*L*(FEAT/4)
    int    f4  = (int)(gid & 127);
    size_t bt  = gid >> 7;
    int    tt  = (int)(bt & (LL - 1));
    int    bat = (int)(bt >> 12);

    f32x4 a = {0.0f, 0.0f, 0.0f, 0.0f};
    #pragma unroll
    for (int k = 0; k < TOPK; ++k) {
        int src = (tt + idx[k]) & (LL - 1);
        f32x4 p = *(const f32x4*)(v + ((size_t)bat * LL + src) * FEAT + f4 * 4);
        a += p * wsm[bat * TOPK + k];
    }
    *(f32x4*)(out + gid * 4) = a;
}

// ---------------------------------------------------------------------------
extern "C" void kernel_launch(void* const* d_in, const int* in_sizes, int n_in,
                              void* d_out, int out_size, void* d_ws, size_t ws_size,
                              hipStream_t stream) {
    const float* q = (const float*)d_in[0];
    const float* k = (const float*)d_in[1];
    const float* v = (const float*)d_in[2];
    float* out = (float*)d_out;

    char* ws = (char*)d_ws;
    const size_t fp16_bytes = (size_t)BB * LL * FEAT * 2;   // 32 MB each
    _Float16* qh   = (_Float16*)ws;
    _Float16* kh   = (_Float16*)(ws + fp16_bytes);
    float*    mean = (float*)(ws + 2 * fp16_bytes);
    int*      idxb = (int*)(ws + 2 * fp16_bytes + (size_t)BB * LL * 4);
    float*    wsm  = (float*)(ws + 2 * fp16_bytes + (size_t)BB * LL * 4 + 64);

    const int n4 = BB * LL * FEAT / 4;                       // 4,194,304
    cvt_f16_kernel<<<n4 / 256, 256, 0, stream>>>(q, qh, n4);
    cvt_f16_kernel<<<n4 / 256, 256, 0, stream>>>(k, kh, n4);
    zero_kernel<<<(BB * LL) / 256, 256, 0, stream>>>(mean, BB * LL);

    gram_diag_kernel<<<BB * 16 * 16, 512, 0, stream>>>(qh, kh, mean);

    topk_softmax_kernel<<<1, 256, 0, stream>>>(mean, idxb, wsm);

    const int ng = BB * LL * (FEAT / 4) / 256;               // 16384
    gather_agg_kernel<<<ng, 256, 0, stream>>>(v, idxb, wsm, out);
}
